// SABlock_9363028705845
// MI455X (gfx1250) — matrix operand order repaired and checked by reading, not compiled, not run on hardware
//
#include <hip/hip_runtime.h>
#include <hip/hip_bf16.h>

typedef _Float16 half_t;
typedef __attribute__((ext_vector_type(16))) _Float16 v16h;
typedef __attribute__((ext_vector_type(8)))  _Float16 v8h;
typedef __attribute__((ext_vector_type(8)))  float    v8f;
typedef __attribute__((ext_vector_type(4)))  int      v4i;

#if defined(__AMDGCN__) && __has_builtin(__builtin_amdgcn_global_load_async_to_lds_b128)
#define HAVE_ASYNC_LDS 1
#else
#define HAVE_ASYNC_LDS 0
#endif

__device__ __forceinline__ v8f vzero8f() {
    v8f z = {0.f, 0.f, 0.f, 0.f, 0.f, 0.f, 0.f, 0.f};
    return z;
}

#if HAVE_ASYNC_LDS
typedef __attribute__((address_space(1))) v4i gv4i;  // global int4
typedef __attribute__((address_space(3))) v4i lv4i;  // LDS int4
// direct global -> LDS async copy of 16 bytes (ASYNCcnt tracked)
__device__ __forceinline__ void async_cp16(const half_t* g, half_t* l) {
    __builtin_amdgcn_global_load_async_to_lds_b128((gv4i*)g, (lv4i*)l, 0, 0);
}
#endif

__device__ __forceinline__ void wait_async0() {
#if __has_builtin(__builtin_amdgcn_s_wait_asynccnt)
    __builtin_amdgcn_s_wait_asynccnt(0);
#elif defined(__AMDGCN__)
    asm volatile("s_wait_asynccnt 0x0" ::: "memory");
#endif
}

// ---------------------------------------------------------------------------
// WMMA fragment loaders (contiguous k order in A and B, wave32)
// A 16x32 (MxK): lane m = lane&15, lane-group (lane>>4) adds K+16.
//   VGPR p: K = kg*16 + 2p (+1 for hi half)
// B 32x16 (KxN): lane n = lane&15, lane-group adds K+16.
//   VGPR p: K = kg*16 + 2p (+1)
// ---------------------------------------------------------------------------
__device__ __forceinline__ v16h frag_a(const half_t* t, int row0, int k0, int ld) {
    int lane = threadIdx.x & 31;
    const half_t* p = t + (row0 + (lane & 15)) * ld + k0 + ((lane >> 4) << 4);
    v16h f;
#pragma unroll
    for (int q = 0; q < 8; ++q) {
        int k = q << 1;
        f[2 * q]     = p[k];
        f[2 * q + 1] = p[k + 1];
    }
    return f;
}

__device__ __forceinline__ v16h frag_b(const half_t* t, int col0, int k0, int ld) {
    // B stored row-major [K][N]
    int lane = threadIdx.x & 31;
    const half_t* p = t + col0 + (lane & 15) + (long)(k0 + ((lane >> 4) << 4)) * ld;
    v16h f;
#pragma unroll
    for (int q = 0; q < 8; ++q) {
        f[2 * q]     = p[(2 * q) * ld];
        f[2 * q + 1] = p[(2 * q + 1) * ld];
    }
    return f;
}

__device__ __forceinline__ v16h frag_bt(const half_t* t, int row0, int k0, int ld) {
    // B fragment taken from a row-major [N][K] matrix (i.e. multiply by t^T):
    // lane n reads K contiguously -> vector load from LDS.
    int lane = threadIdx.x & 31;
    const half_t* p = t + (row0 + (lane & 15)) * ld + k0 + ((lane >> 4) << 4);
    v16h f;
#pragma unroll
    for (int q = 0; q < 16; ++q) f[q] = p[q];
    return f;
}

__device__ __forceinline__ v8f wmma_f16(v16h a, v16h b, v8f c) {
    return __builtin_amdgcn_wmma_f32_16x16x32_f16(false, a, false, b, (short)0, c,
                                                  false, false);
}

// ---------------------------------------------------------------------------
// fp32 -> f16 convert
// ---------------------------------------------------------------------------
__global__ void k_cvt(const float* __restrict__ a, half_t* __restrict__ o, int n) {
    int i = blockIdx.x * 256 + threadIdx.x;
    if (i < n) o[i] = (half_t)a[i];
}

// ---------------------------------------------------------------------------
// Window gather + LayerNorm1 -> f16 [window-major rows, 384]
// ---------------------------------------------------------------------------
__global__ __launch_bounds__(256) void k_gather_ln1(
    const float* __restrict__ x, const float* __restrict__ g,
    const float* __restrict__ bta, half_t* __restrict__ out) {
    int wave = threadIdx.x >> 5, lane = threadIdx.x & 31;
    long r   = (long)blockIdx.x * 8 + wave;       // 0 .. 100351 window-major
    int win  = (int)(r >> 6), t = (int)(r & 63);
    int b = win / 49, rem = win % 49, gy = rem / 7, gx = rem % 7;
    int sy = t >> 3, sx = t & 7;
    int h = sy * 7 + gy, w = sx * 7 + gx;
    const int HW = 56 * 56;
    float v[12];
    float s = 0.f, ss = 0.f;
#pragma unroll
    for (int j = 0; j < 12; ++j) {
        int c    = lane + (j << 5);
        float xv = x[((long)b * 384 + c) * HW + h * 56 + w];
        v[j] = xv; s += xv; ss += xv * xv;
    }
#pragma unroll
    for (int o = 16; o > 0; o >>= 1) {
        s += __shfl_xor(s, o, 32);
        ss += __shfl_xor(ss, o, 32);
    }
    float mean = s * (1.f / 384.f);
    float var  = ss * (1.f / 384.f) - mean * mean;
    float inv  = rsqrtf(var + 1e-5f);
#pragma unroll
    for (int j = 0; j < 12; ++j) {
        int c = lane + (j << 5);
        out[r * 384 + c] = (half_t)((v[j] - mean) * inv * g[c] + bta[c]);
    }
}

// ---------------------------------------------------------------------------
// Tiled WMMA GEMM: C[M,N] = A[M,K] * B[K,N] (+bias) with epilogue modes.
// BM=BN=128, BK=32; 8 waves, each 64x32 subtile (4x2 wmma accumulators).
// Tile staging uses GLOBAL_LOAD_ASYNC_TO_LDS_B128 when available.
// MODE 0: store f16 row-major. MODE 1: store f32 row-major.
// MODE 2: out = NCHW f32: x1 + gamma2*(acc+bias)   (final residual)
// ---------------------------------------------------------------------------
template <int MODE, bool HAS_BIAS>
__global__ __launch_bounds__(256) void k_gemm(
    const half_t* __restrict__ A, const half_t* __restrict__ B,
    const float* __restrict__ bias, void* __restrict__ outp, int M, int N, int K,
    const float* __restrict__ x1, const float* __restrict__ gamma2) {
    constexpr int LDA = 40, LDB = 136;
    __shared__ half_t As[2][128 * LDA];
    __shared__ half_t Bs[2][32 * LDB];
    int tid = threadIdx.x;
    int wave = tid >> 5, lane = tid & 31;
    int wm = (wave >> 2) * 64, wn = (wave & 3) * 32;
    long bm = (long)blockIdx.y * 128;
    long bn = (long)blockIdx.x * 128;

    v8f acc[4][2];
#pragma unroll
    for (int mi = 0; mi < 4; ++mi)
#pragma unroll
        for (int ni = 0; ni < 2; ++ni) acc[mi][ni] = vzero8f();

    auto compute = [&](int cur) {
        v16h af[4], bf[2];
#pragma unroll
        for (int mi = 0; mi < 4; ++mi) af[mi] = frag_a(&As[cur][0], wm + mi * 16, 0, LDA);
#pragma unroll
        for (int ni = 0; ni < 2; ++ni) bf[ni] = frag_b(&Bs[cur][0], wn + ni * 16, 0, LDB);
#pragma unroll
        for (int mi = 0; mi < 4; ++mi)
#pragma unroll
            for (int ni = 0; ni < 2; ++ni)
                acc[mi][ni] = wmma_f16(af[mi], bf[ni], acc[mi][ni]);
    };

    int nk = K >> 5;
#if HAVE_ASYNC_LDS
    auto aload = [&](int buf, int k0) {
#pragma unroll
        for (int j = 0; j < 2; ++j) {
            int c = tid + 256 * j;                 // 0..511 chunks of 8 halves
            int arow = c >> 2, aoff = (c & 3) << 3;
            async_cp16(A + (bm + arow) * K + k0 + aoff, &As[buf][arow * LDA + aoff]);
            int brow = c >> 4, boff = (c & 15) << 3;
            async_cp16(B + (long)(k0 + brow) * N + bn + boff, &Bs[buf][brow * LDB + boff]);
        }
    };
    aload(0, 0);
    wait_async0();
    __syncthreads();
    for (int kt = 0; kt < nk; ++kt) {
        if (kt + 1 < nk) aload((kt + 1) & 1, (kt + 1) << 5);
        compute(kt & 1);
        if (kt + 1 < nk) wait_async0();
        __syncthreads();
    }
#else
    v8h ra[2], rb[2];
    auto gload = [&](int k0) {
#pragma unroll
        for (int j = 0; j < 2; ++j) {
            int c = tid + 256 * j;
            int arow = c >> 2, aoff = (c & 3) << 3;
            ra[j] = *(const v8h*)(A + (bm + arow) * K + k0 + aoff);
            int brow = c >> 4, boff = (c & 15) << 3;
            rb[j] = *(const v8h*)(B + (long)(k0 + brow) * N + bn + boff);
        }
    };
    auto lstore = [&](int buf) {
#pragma unroll
        for (int j = 0; j < 2; ++j) {
            int c = tid + 256 * j;
            int arow = c >> 2, aoff = (c & 3) << 3;
            *(v8h*)&As[buf][arow * LDA + aoff] = ra[j];
            int brow = c >> 4, boff = (c & 15) << 3;
            *(v8h*)&Bs[buf][brow * LDB + boff] = rb[j];
        }
    };
    gload(0);
    lstore(0);
    __syncthreads();
    for (int kt = 0; kt < nk; ++kt) {
        if (kt + 1 < nk) gload((kt + 1) << 5);
        compute(kt & 1);
        if (kt + 1 < nk) lstore((kt + 1) & 1);
        __syncthreads();
    }
#endif

#pragma unroll
    for (int mi = 0; mi < 4; ++mi)
#pragma unroll
        for (int ni = 0; ni < 2; ++ni) {
            long gm0 = bm + wm + mi * 16 + ((lane >> 4) << 3);
            int gn   = (int)bn + wn + ni * 16 + (lane & 15);
            float bv = HAS_BIAS ? bias[gn] : 0.f;
#pragma unroll
            for (int i = 0; i < 8; ++i) {
                long gm = gm0 + i;
                float v = acc[mi][ni][i] + bv;
                if (MODE == 0) {
                    ((half_t*)outp)[gm * N + gn] = (half_t)v;
                } else if (MODE == 1) {
                    ((float*)outp)[gm * N + gn] = v;
                } else {
                    long b = gm / 3136, n = gm % 3136;
                    float r = x1[gm * N + gn] + gamma2[gn] * v;
                    ((float*)outp)[(b * 384 + gn) * 3136 + n] = r;
                }
            }
        }
}

// ---------------------------------------------------------------------------
// Window attention: one wave per (window, head). n=64 tokens, d=48 (pad->64).
// S = Q K^T (WMMA, frag_bt), softmax in LDS, O = P V (WMMA, frag_b).
// ---------------------------------------------------------------------------
__global__ __launch_bounds__(32) void k_attn(const half_t* __restrict__ qkv,
                                             half_t* __restrict__ o) {
    __shared__ half_t Qs[64 * 64];
    __shared__ half_t Ks[64 * 64];
    __shared__ half_t Vs[64 * 64];
    __shared__ float  Ss[64 * 64];
    __shared__ half_t Ps[64 * 64];
    const int LD = 64;
    int lane = threadIdx.x;
    int win  = blockIdx.x;
    int head = blockIdx.y;

    v8h hz;
#pragma unroll
    for (int e = 0; e < 8; ++e) hz[e] = (half_t)0.f;

    for (int r = lane; r < 64; r += 32) {
        long base = ((long)win * 64 + r) * 1152 + head * 48;
#pragma unroll
        for (int j = 0; j < 6; ++j) {
#if HAVE_ASYNC_LDS
            async_cp16(qkv + base + j * 8,       &Qs[r * LD + j * 8]);
            async_cp16(qkv + base + 384 + j * 8, &Ks[r * LD + j * 8]);
            async_cp16(qkv + base + 768 + j * 8, &Vs[r * LD + j * 8]);
#else
            *(v8h*)&Qs[r * LD + j * 8] = *(const v8h*)(qkv + base + j * 8);
            *(v8h*)&Ks[r * LD + j * 8] = *(const v8h*)(qkv + base + 384 + j * 8);
            *(v8h*)&Vs[r * LD + j * 8] = *(const v8h*)(qkv + base + 768 + j * 8);
#endif
        }
#pragma unroll
        for (int j = 6; j < 8; ++j) {
            *(v8h*)&Qs[r * LD + j * 8] = hz;
            *(v8h*)&Ks[r * LD + j * 8] = hz;
            *(v8h*)&Vs[r * LD + j * 8] = hz;
        }
    }
#if HAVE_ASYNC_LDS
    wait_async0();
#endif
    __syncthreads();

    // S = Q * K^T   (64x64, k=64 padded)
#pragma unroll
    for (int mi = 0; mi < 4; ++mi)
#pragma unroll
        for (int ni = 0; ni < 4; ++ni) {
            v8f acc = vzero8f();
#pragma unroll
            for (int kk = 0; kk < 2; ++kk) {
                v16h a = frag_a(Qs, mi * 16, kk * 32, LD);
                v16h b = frag_bt(Ks, ni * 16, kk * 32, LD);
                acc = wmma_f16(a, b, acc);
            }
            int n = ni * 16 + (lane & 15);
#pragma unroll
            for (int i = 0; i < 8; ++i) {
                int m = mi * 16 + i + ((lane & 16) ? 8 : 0);
                Ss[m * LD + n] = acc[i];
            }
        }
    __syncthreads();

    // softmax(scale * S) rows
    const float scale = 0.14433756729740643f;  // 48^-0.5
    for (int r = lane; r < 64; r += 32) {
        float mx = -1e30f;
        for (int j = 0; j < 64; ++j) mx = fmaxf(mx, Ss[r * LD + j]);
        float sum = 0.f;
        for (int j = 0; j < 64; ++j) {
            float e = __expf((Ss[r * LD + j] - mx) * scale);
            Ss[r * LD + j] = e;
            sum += e;
        }
        float inv = 1.f / sum;
        for (int j = 0; j < 64; ++j) Ps[r * LD + j] = (half_t)(Ss[r * LD + j] * inv);
    }
    __syncthreads();

    // O = P * V   (64x48, k=64)
#pragma unroll
    for (int mi = 0; mi < 4; ++mi)
#pragma unroll
        for (int nd = 0; nd < 3; ++nd) {
            v8f acc = vzero8f();
#pragma unroll
            for (int kk = 0; kk < 2; ++kk) {
                v16h a = frag_a(Ps, mi * 16, kk * 32, LD);
                v16h b = frag_b(Vs, nd * 16, kk * 32, LD);
                acc = wmma_f16(a, b, acc);
            }
            int n = head * 48 + nd * 16 + (lane & 15);
#pragma unroll
            for (int i = 0; i < 8; ++i) {
                int m = mi * 16 + i + ((lane & 16) ? 8 : 0);
                o[((long)win * 64 + m) * 384 + n] = (half_t)acc[i];
            }
        }
}

// ---------------------------------------------------------------------------
// Unsparse scatter + residual (gamma1) + LayerNorm2.
// ---------------------------------------------------------------------------
__global__ __launch_bounds__(256) void k_scatter_ln2(
    const float* __restrict__ x, const float* __restrict__ po,
    const float* __restrict__ gamma1, const float* __restrict__ g2,
    const float* __restrict__ b2, float* __restrict__ x1,
    half_t* __restrict__ h2) {
    int wave = threadIdx.x >> 5, lane = threadIdx.x & 31;
    long row = (long)blockIdx.x * 8 + wave;  // token-major b*3136+n
    int b = (int)(row / 3136), n = (int)(row % 3136);
    int h = n / 56, w = n % 56;
    int sy = h / 7, gy = h % 7, sx = w / 7, gx = w % 7;
    long rw = ((long)(b * 49 + gy * 7 + gx)) * 64 + sy * 8 + sx;
    float v[12];
    float s = 0.f, ss = 0.f;
#pragma unroll
    for (int j = 0; j < 12; ++j) {
        int c = lane + (j << 5);
        float xv  = x[((long)b * 384 + c) * 3136 + n];
        float val = xv + gamma1[c] * po[rw * 384 + c];
        v[j] = val;
        x1[row * 384 + c] = val;
        s += val; ss += val * val;
    }
#pragma unroll
    for (int o = 16; o > 0; o >>= 1) {
        s += __shfl_xor(s, o, 32);
        ss += __shfl_xor(ss, o, 32);
    }
    float mean = s * (1.f / 384.f);
    float var  = ss * (1.f / 384.f) - mean * mean;
    float inv  = rsqrtf(var + 1e-5f);
#pragma unroll
    for (int j = 0; j < 12; ++j) {
        int c = lane + (j << 5);
        h2[row * 384 + c] = (half_t)((v[j] - mean) * inv * g2[c] + b2[c]);
    }
}

// ---------------------------------------------------------------------------
// Depthwise 3x3 conv (+bias) + exact GELU; 8 channels per thread, f16 io.
// ---------------------------------------------------------------------------
__global__ __launch_bounds__(256) void k_dwgelu(const half_t* __restrict__ m1,
                                                const float* __restrict__ dww,
                                                const float* __restrict__ dwb,
                                                half_t* __restrict__ act) {
    long gid  = (long)blockIdx.x * 256 + threadIdx.x;
    int token = (int)(gid / 192);
    int cb    = (int)(gid % 192) * 8;
    int b = token / 3136, n = token % 3136;
    int h = n / 56, w = n % 56;
    float acc[8];
#pragma unroll
    for (int e = 0; e < 8; ++e) acc[e] = dwb[cb + e];
#pragma unroll
    for (int dy = -1; dy <= 1; ++dy) {
        int hh = h + dy;
        if (hh < 0 || hh >= 56) continue;
#pragma unroll
        for (int dx = -1; dx <= 1; ++dx) {
            int ww = w + dx;
            if (ww < 0 || ww >= 56) continue;
            const half_t* p = m1 + ((long)b * 3136 + hh * 56 + ww) * 1536 + cb;
            v8h hv = *(const v8h*)p;
#pragma unroll
            for (int e = 0; e < 8; ++e)
                acc[e] += (float)hv[e] * dww[(cb + e) * 9 + (dy + 1) * 3 + (dx + 1)];
        }
    }
    v8h ov;
#pragma unroll
    for (int e = 0; e < 8; ++e) {
        float xv = acc[e];
        ov[e] = (half_t)(0.5f * xv * (1.f + erff(xv * 0.70710678118f)));
    }
    *(v8h*)(act + (long)token * 1536 + cb) = ov;
}

// ---------------------------------------------------------------------------
// Launch
// ---------------------------------------------------------------------------
extern "C" void kernel_launch(void* const* d_in, const int* in_sizes, int n_in,
                              void* d_out, int out_size, void* d_ws, size_t ws_size,
                              hipStream_t stream) {
    (void)in_sizes; (void)n_in; (void)out_size; (void)ws_size;
    const float* x      = (const float*)d_in[0];
    const float* ln1_g  = (const float*)d_in[1];
    const float* ln1_b  = (const float*)d_in[2];
    const float* qkv_w  = (const float*)d_in[3];
    const float* proj_w = (const float*)d_in[4];
    const float* proj_b = (const float*)d_in[5];
    const float* ln2_g  = (const float*)d_in[6];
    const float* ln2_b  = (const float*)d_in[7];
    const float* fc1_w  = (const float*)d_in[8];
    const float* fc1_b  = (const float*)d_in[9];
    const float* dw_w   = (const float*)d_in[10];
    const float* dw_b   = (const float*)d_in[11];
    const float* fc2_w  = (const float*)d_in[12];
    const float* fc2_b  = (const float*)d_in[13];
    const float* gamma1 = (const float*)d_in[14];
    const float* gamma2 = (const float*)d_in[15];
    float* out = (float*)d_out;

    const long M = 100352;  // B*N tokens
    size_t off = 0;
    auto alloc = [&](size_t bytes) {
        void* p = (char*)d_ws + off;
        off += (bytes + 255) & ~(size_t)255;
        return p;
    };
    half_t* wq  = (half_t*)alloc(442368 * 2);          // [384,1152]
    half_t* wp  = (half_t*)alloc(147456 * 2);          // [384,384]
    half_t* w1  = (half_t*)alloc(589824 * 2);          // [384,1536]
    half_t* w2  = (half_t*)alloc(589824 * 2);          // [1536,384]
    half_t* win = (half_t*)alloc((size_t)M * 384 * 2); // xw_ln / attn_o / h2
    half_t* r1  = (half_t*)alloc((size_t)M * 1536 * 2);// qkv then m1
    void*   r2  = alloc((size_t)M * 1536 * 2);         // proj_o(f32) then act(f16)
    float*  x1  = (float*)alloc((size_t)M * 384 * 4);

    // weights -> f16
    k_cvt<<<(442368 + 255) / 256, 256, 0, stream>>>(qkv_w, wq, 442368);
    k_cvt<<<(147456 + 255) / 256, 256, 0, stream>>>(proj_w, wp, 147456);
    k_cvt<<<(589824 + 255) / 256, 256, 0, stream>>>(fc1_w, w1, 589824);
    k_cvt<<<(589824 + 255) / 256, 256, 0, stream>>>(fc2_w, w2, 589824);

    // gather + LN1
    k_gather_ln1<<<12544, 256, 0, stream>>>(x, ln1_g, ln1_b, win);

    // QKV GEMM: [M,384] x [384,1152] -> f16
    k_gemm<0, false><<<dim3(1152 / 128, 784), 256, 0, stream>>>(
        win, wq, nullptr, r1, (int)M, 1152, 384, nullptr, nullptr);

    // window attention (attn_o reuses `win`)
    k_attn<<<dim3(1568, 8), 32, 0, stream>>>(r1, win);

    // proj GEMM: [M,384] x [384,384] + bias -> f32
    k_gemm<1, true><<<dim3(384 / 128, 784), 256, 0, stream>>>(
        win, wp, proj_b, r2, (int)M, 384, 384, nullptr, nullptr);

    // unsparse + residual + LN2 (h2 reuses `win`)
    k_scatter_ln2<<<12544, 256, 0, stream>>>(x, (const float*)r2, gamma1, ln2_g,
                                             ln2_b, x1, win);

    // FC1 GEMM: [M,384] x [384,1536] + bias -> f16 (m1 in r1)
    k_gemm<0, true><<<dim3(1536 / 128, 784), 256, 0, stream>>>(
        win, w1, fc1_b, r1, (int)M, 1536, 384, nullptr, nullptr);

    // depthwise conv + gelu -> f16 (act in r2)
    k_dwgelu<<<75264, 256, 0, stream>>>(r1, dw_w, dw_b, (half_t*)r2);

    // FC2 GEMM + final residual -> NCHW f32 out
    k_gemm<2, true><<<dim3(384 / 128, 784), 256, 0, stream>>>(
        (const half_t*)r2, w2, fc2_b, out, (int)M, 384, 1536, x1, gamma2);
}
